// AttnWeightRoILocalizer_79757542687322
// MI455X (gfx1250) — matrix operand order repaired and checked by reading, not compiled, not run on hardware
//
#include <hip/hip_runtime.h>
#include <hip/hip_bf16.h>

// ---- problem constants (match reference) ----
#define B_      16
#define C_IN    3
#define IMG     512
#define PATCH   16
#define C_FEAT  2048
#define N_CLS   18
#define K_CELLS 24
#define HF      32
#define P_      (HF * HF)        // 1024 patches per image
#define M_PATCH (B_ * P_)        // 16384 total patches
#define K_BB    (C_IN * PATCH * PATCH) // 768 im2col depth
#define H_DIM   1024
#define EPSF    1e-6f

// ---- WMMA vector types (CDNA5 wave32) ----
typedef __attribute__((ext_vector_type(8)))  __bf16 v8bf;
typedef __attribute__((ext_vector_type(16))) __bf16 v16bf;
typedef __attribute__((ext_vector_type(8)))  float  v8f;

__device__ inline __bf16 f2bf(float f) {
  unsigned u = __builtin_bit_cast(unsigned, f);
  unsigned r = u + 0x7FFFu + ((u >> 16) & 1u);   // round-to-nearest-even
  return __builtin_bit_cast(__bf16, (unsigned short)(r >> 16));
}
__device__ inline float bf2f(__bf16 b) {
  unsigned u = ((unsigned)__builtin_bit_cast(unsigned short, b)) << 16;
  return __builtin_bit_cast(float, u);
}

// A fragment 16x32 (MxK), A row-major with leading dim lda (elements).
// Layout used here (same as B): lane row = lane%16; 16 contiguous K starting at (lane/16)*16
__device__ inline v16bf load_a_frag(const __bf16* __restrict__ A, int lda, int lane) {
  const __bf16* p = A + (size_t)(lane & 15) * lda + ((lane >> 4) << 4);
  v8bf lo = *(const v8bf*)(p);
  v8bf hi = *(const v8bf*)(p + 8);
  return __builtin_shufflevector(lo, hi, 0,1,2,3,4,5,6,7,8,9,10,11,12,13,14,15);
}
// B fragment 32x16 (KxN), B stored as [N][K] row-major with leading dim ldb.
// Layout used here (same as A): lane col = lane%16; 16 contiguous K starting at (lane/16)*16
__device__ inline v16bf load_b_frag(const __bf16* __restrict__ Bt, int ldb, int lane) {
  const __bf16* p = Bt + (size_t)(lane & 15) * ldb + ((lane >> 4) << 4);
  return *(const v16bf*)(p);
}
__device__ inline v8f wmma_bf16(v16bf a, v16bf b, v8f c) {
  return __builtin_amdgcn_wmma_f32_16x16x32_bf16(false, a, false, b, (short)0, c, false, false);
}

// ---------------- elementwise prep ----------------
__global__ void f32_to_bf16_k(const float* __restrict__ src, __bf16* __restrict__ dst, int n) {
  int i = blockIdx.x * blockDim.x + threadIdx.x;
  if (i < n) dst[i] = f2bf(src[i]);
}

// im2col: A[m][k], m = b*1024 + py*32 + px, k = c*256 + y*16 + x
__global__ void im2col_k(const float* __restrict__ img, __bf16* __restrict__ A) {
  size_t idx = (size_t)blockIdx.x * blockDim.x + threadIdx.x;
  if (idx >= (size_t)M_PATCH * K_BB) return;
  int k = (int)(idx % K_BB);
  int m = (int)(idx / K_BB);
  int b  = m >> 10, pp = m & 1023;
  int py = pp >> 5, px = pp & 31;
  int c  = k >> 8,  rr = k & 255;
  int y  = rr >> 4, x  = rr & 15;
  size_t src = (((size_t)b * C_IN + c) * IMG + (size_t)(py * PATCH + y)) * IMG
             + (size_t)(px * PATCH + x);
  A[idx] = f2bf(img[src]);
}

// masks -> bf16 A-matrix [B][32][1024] (rows 24..31 zero) + raw areas S[384]
__global__ void maskprep_k(const float* __restrict__ masks, __bf16* __restrict__ Mb,
                           float* __restrict__ S) {
  __shared__ float red[256];
  int bk = blockIdx.x;          // 0..511
  int b = bk >> 5, row = bk & 31;
  __bf16* dst = Mb + ((size_t)b * 32 + row) * P_;
  if (row < K_CELLS) {
    const float* src = masks + ((size_t)b * K_CELLS + row) * P_;
    float psum = 0.f;
    for (int i = threadIdx.x; i < P_; i += 256) {
      float v = src[i];
      dst[i] = f2bf(v);
      psum += v;
    }
    red[threadIdx.x] = psum;
    __syncthreads();
    for (int s = 128; s > 0; s >>= 1) {
      if ((int)threadIdx.x < s) red[threadIdx.x] += red[threadIdx.x + s];
      __syncthreads();
    }
    if (threadIdx.x == 0) S[b * K_CELLS + row] = red[0];
  } else {
    for (int i = threadIdx.x; i < P_; i += 256) dst[i] = f2bf(0.f);
  }
}

// ---------------- GEMM 1: patch embed, M=16384 N=2048 K=768 ----------------
// Block tile 128x128, LDS double-buffered; 8 waves, each computes 32x64 (8 WMMA/step).
// LDS rows padded to 40 elems (20 banks) -> conflict-free fragment reads.
#define BM  128
#define BN  128
#define BK  32
#define BKP 40   // padded row length in elements (80 B = 20 banks)

__global__ __launch_bounds__(256) void gemm_backbone_k(const __bf16* __restrict__ A,
                                                       const __bf16* __restrict__ Bw,
                                                       const float* __restrict__ bias,
                                                       __bf16* __restrict__ fmT) {
  __shared__ __align__(32) __bf16 sA[2][BM * BKP];   // 10 KB per buffer
  __shared__ __align__(32) __bf16 sB[2][BN * BKP];
  const int tid = threadIdx.x;
  const int lane = tid & 31, wave = tid >> 5;
  const int wm = wave & 3;        // 0..3 -> 32-row strip
  const int wn = wave >> 2;       // 0..1 -> 64-col strip
  const int m0 = blockIdx.y * BM;
  const int n0 = blockIdx.x * BN;

  v8f acc[2][4] = {};

  // cooperative stage of one 128x32 A tile + 128x32 B tile (16-byte chunks)
  auto stage = [&](int k0, int buf) {
#pragma unroll
    for (int q = 0; q < 2; ++q) {
      int c = tid + q * 256;                 // 0..511 chunk id
      int row = c >> 2, j = c & 3;           // 4 chunks of 8 elems per row
      __builtin_prefetch(A + (size_t)(m0 + row) * K_BB + k0 + 256, 0, 1);
      *(v8bf*)(&sA[buf][row * BKP + j * 8]) =
          *(const v8bf*)(A + (size_t)(m0 + row) * K_BB + k0 + j * 8);
      *(v8bf*)(&sB[buf][row * BKP + j * 8]) =
          *(const v8bf*)(Bw + (size_t)(n0 + row) * K_BB + k0 + j * 8);
    }
  };
  auto a_frag = [&](int buf, int mt) -> v16bf {
    int r = wm * 32 + mt * 16 + (lane & 15);
    int kb = (lane >> 4) << 4;
    v8bf lo = *(const v8bf*)(&sA[buf][r * BKP + kb]);
    v8bf hi = *(const v8bf*)(&sA[buf][r * BKP + kb + 8]);
    return __builtin_shufflevector(lo, hi, 0,1,2,3,4,5,6,7,8,9,10,11,12,13,14,15);
  };
  auto b_frag = [&](int buf, int nt) -> v16bf {
    int n = wn * 64 + nt * 16 + (lane & 15);
    int kb = (lane >> 4) << 4;
    v8bf lo = *(const v8bf*)(&sB[buf][n * BKP + kb]);
    v8bf hi = *(const v8bf*)(&sB[buf][n * BKP + kb + 8]);
    return __builtin_shufflevector(lo, hi, 0,1,2,3,4,5,6,7,8,9,10,11,12,13,14,15);
  };

  stage(0, 0);
  __syncthreads();
  const int nsteps = K_BB / BK;   // 24
  for (int s = 0; s < nsteps; ++s) {
    const int buf = s & 1;
    if (s + 1 < nsteps) stage((s + 1) * BK, buf ^ 1);   // overlap next tile
    v16bf a0 = a_frag(buf, 0);
    v16bf a1 = a_frag(buf, 1);
#pragma unroll
    for (int t = 0; t < 4; ++t) {
      v16bf bb = b_frag(buf, t);
      acc[0][t] = wmma_bf16(a0, bb, acc[0][t]);
      acc[1][t] = wmma_bf16(a1, bb, acc[1][t]);
    }
    __syncthreads();
  }

  const int col = lane & 15, radd = (lane >> 4) << 3;
#pragma unroll
  for (int mt = 0; mt < 2; ++mt) {
#pragma unroll
    for (int t = 0; t < 4; ++t) {
      const int n = n0 + wn * 64 + t * 16 + col;
      const float bv = bias[n];
#pragma unroll
      for (int r = 0; r < 8; ++r) {
        const int m = m0 + wm * 32 + mt * 16 + r + radd;
        float v = acc[mt][t][r] + bv;
        v = v > 0.f ? v : 0.f;
        fmT[(size_t)n * M_PATCH + m] = f2bf(v);  // transposed: [feat][patch]
      }
    }
  }
}

// ---------------- GEMM 2: RoI pooling, per image M=32 N=2048 K=1024 -------
// grid (2048/256, B), 8 waves: wave&1 -> m-tile, wave>>1 -> 64-wide n strip
__global__ __launch_bounds__(256) void gemm_roi_k(const __bf16* __restrict__ Mb,
                                                  const __bf16* __restrict__ fmT,
                                                  const float* __restrict__ S,
                                                  float* __restrict__ cf32,
                                                  __bf16* __restrict__ cf16) {
  const int lane = threadIdx.x & 31, wave = threadIdx.x >> 5;
  const int b = blockIdx.y;
  const int m0 = (wave & 1) * 16;
  const int n0 = blockIdx.x * 256 + (wave >> 1) * 64;
  v8f acc[4] = {};
  const __bf16* aBase = Mb + ((size_t)b * 32 + m0) * P_;
  for (int k0 = 0; k0 < P_; k0 += 32) {
    v16bf a = load_a_frag(aBase + k0, P_, lane);
#pragma unroll
    for (int t = 0; t < 4; ++t) {
      v16bf bb = load_b_frag(fmT + (size_t)(n0 + t * 16) * M_PATCH + (size_t)b * P_ + k0,
                             M_PATCH, lane);
      acc[t] = wmma_bf16(a, bb, acc[t]);
    }
  }
  const int col = lane & 15, radd = (lane >> 4) << 3;
#pragma unroll
  for (int t = 0; t < 4; ++t) {
    const int n = n0 + t * 16 + col;
#pragma unroll
    for (int r = 0; r < 8; ++r) {
      const int cr = m0 + r + radd;
      if (cr < K_CELLS) {
        const int cell = b * K_CELLS + cr;
        const float v = acc[t][r] / (S[cell] + EPSF);
        cf32[(size_t)cell * C_FEAT + n] = v;
        cf16[(size_t)cell * C_FEAT + n] = f2bf(v);
      }
    }
  }
}

// ---------------- GEMM 3: MLP fc1, M=384 N=1024 K=2048 --------------------
__global__ __launch_bounds__(256) void gemm_mlp_k(const __bf16* __restrict__ A,
                                                  const __bf16* __restrict__ Bw,
                                                  const float* __restrict__ bias,
                                                  __bf16* __restrict__ H) {
  const int lane = threadIdx.x & 31, wave = threadIdx.x >> 5;
  const int m0 = blockIdx.y * 128 + wave * 16;
  const int n0 = blockIdx.x * 64;
  v8f acc[4] = {};
  const __bf16* aBase = A + (size_t)m0 * C_FEAT;
  for (int k0 = 0; k0 < C_FEAT; k0 += 32) {
    __builtin_prefetch(aBase + k0 + 256, 0, 1);
    v16bf a = load_a_frag(aBase + k0, C_FEAT, lane);
#pragma unroll
    for (int t = 0; t < 4; ++t) {
      v16bf bb = load_b_frag(Bw + (size_t)(n0 + t * 16) * C_FEAT + k0, C_FEAT, lane);
      acc[t] = wmma_bf16(a, bb, acc[t]);
    }
  }
  const int col = lane & 15, radd = (lane >> 4) << 3;
#pragma unroll
  for (int t = 0; t < 4; ++t) {
    const int n = n0 + t * 16 + col;
    const float bv = bias[n];
#pragma unroll
    for (int r = 0; r < 8; ++r) {
      const int m = m0 + r + radd;
      float v = acc[t][r] + bv;
      v = v > 0.f ? v : 0.f;
      H[(size_t)m * H_DIM + n] = f2bf(v);
    }
  }
}

// class logits per cell: cf32 . W_final^T + b_final * S/(S+eps)
__global__ void class_logits_k(const float* __restrict__ cf32, const float* __restrict__ S,
                               const float* __restrict__ Wf, const float* __restrict__ bfin,
                               float* __restrict__ cl) {
  int cell = blockIdx.x, o = threadIdx.x;
  if (o >= N_CLS) return;
  const float* feat = cf32 + (size_t)cell * C_FEAT;
  const float* w = Wf + (size_t)o * C_FEAT;
  float acc = 0.f;
  for (int c = 0; c < C_FEAT; ++c) acc += feat[c] * w[c];
  float s = S[cell];
  cl[cell * N_CLS + o] = acc + bfin[o] * (s / (s + EPSF));
}

// weight logits per cell: h . W2^T + b2
__global__ void weight_logits_k(const __bf16* __restrict__ H, const float* __restrict__ W2,
                                const float* __restrict__ b2, float* __restrict__ wl) {
  int cell = blockIdx.x, o = threadIdx.x;
  if (o >= N_CLS) return;
  const __bf16* hp = H + (size_t)cell * H_DIM;
  const float* w = W2 + (size_t)o * H_DIM;
  float acc = 0.f;
  for (int i = 0; i < H_DIM; ++i) acc += bf2f(hp[i]) * w[i];
  wl[cell * N_CLS + o] = acc + b2[o];
}

// softmax over cells (per class) + weighted sum; fallback = spatial max of class maps
__global__ void finalize_k(const float* __restrict__ wl, const float* __restrict__ cl,
                           const int* __restrict__ counts, const float* __restrict__ Wf,
                           const float* __restrict__ bfin, const __bf16* __restrict__ fmT,
                           float* __restrict__ out) {
  int b = blockIdx.x, o = threadIdx.x;
  if (o >= N_CLS) return;
  if (counts[b] > 0) {
    float mx = -3.4e38f;
    for (int k = 0; k < K_CELLS; ++k)
      mx = fmaxf(mx, wl[(b * K_CELLS + k) * N_CLS + o]);
    float den = 0.f, num = 0.f;
    for (int k = 0; k < K_CELLS; ++k) {
      float e = expf(wl[(b * K_CELLS + k) * N_CLS + o] - mx);
      den += e;
      num += e * cl[(b * K_CELLS + k) * N_CLS + o];
    }
    out[b * N_CLS + o] = num / den;
  } else {
    // never taken with these inputs (counts==24); kept for generality
    float mx = -3.4e38f;
    for (int p = 0; p < P_; ++p) {
      float v = bfin[o];
      for (int c = 0; c < C_FEAT; ++c)
        v += Wf[o * C_FEAT + c] * bf2f(fmT[(size_t)c * M_PATCH + (size_t)b * P_ + p]);
      mx = fmaxf(mx, v);
    }
    out[b * N_CLS + o] = mx;
  }
}

extern "C" void kernel_launch(void* const* d_in, const int* in_sizes, int n_in,
                              void* d_out, int out_size, void* d_ws, size_t ws_size,
                              hipStream_t stream) {
  (void)in_sizes; (void)n_in; (void)out_size;
  const float* cell_img    = (const float*)d_in[0];
  const float* cell_masks  = (const float*)d_in[1];
  const int*   cell_counts = (const int*)d_in[2];
  const float* W_backbone  = (const float*)d_in[3];
  const float* b_backbone  = (const float*)d_in[4];
  const float* W_final     = (const float*)d_in[5];
  const float* b_final     = (const float*)d_in[6];
  const float* W1f         = (const float*)d_in[7];
  const float* b1          = (const float*)d_in[8];
  const float* W2f         = (const float*)d_in[9];
  const float* b2          = (const float*)d_in[10];
  float* out = (float*)d_out;

  char* w = (char*)d_ws;
  size_t off = 0;
  auto alloc = [&](size_t bytes) -> void* {
    void* p = w + off;
    off = (off + bytes + 255) & ~(size_t)255;
    return p;
  };
  __bf16* A_im2col = (__bf16*)alloc((size_t)M_PATCH * K_BB * 2);   // 25.2 MB
  __bf16* Wb_bf    = (__bf16*)alloc((size_t)C_FEAT * K_BB * 2);    //  3.1 MB
  __bf16* W1_bf    = (__bf16*)alloc((size_t)H_DIM * C_FEAT * 2);   //  4.2 MB
  __bf16* fmT      = (__bf16*)alloc((size_t)C_FEAT * M_PATCH * 2); // 67.1 MB
  __bf16* Mb       = (__bf16*)alloc((size_t)B_ * 32 * P_ * 2);     //  1.0 MB
  float*  S        = (float*) alloc((size_t)B_ * K_CELLS * 4);
  float*  cf32     = (float*) alloc((size_t)B_ * K_CELLS * C_FEAT * 4);
  __bf16* cf16     = (__bf16*)alloc((size_t)B_ * K_CELLS * C_FEAT * 2);
  __bf16* H        = (__bf16*)alloc((size_t)B_ * K_CELLS * H_DIM * 2);
  float*  wl       = (float*) alloc((size_t)B_ * K_CELLS * N_CLS * 4);
  float*  cl       = (float*) alloc((size_t)B_ * K_CELLS * N_CLS * 4);
  if (off > ws_size) return; // workspace too small; nothing safe to do

  // weight conversion
  f32_to_bf16_k<<<(C_FEAT * K_BB + 255) / 256, 256, 0, stream>>>(W_backbone, Wb_bf, C_FEAT * K_BB);
  f32_to_bf16_k<<<(H_DIM * C_FEAT + 255) / 256, 256, 0, stream>>>(W1f, W1_bf, H_DIM * C_FEAT);
  // im2col + mask prep
  im2col_k<<<(int)(((size_t)M_PATCH * K_BB + 255) / 256), 256, 0, stream>>>(cell_img, A_im2col);
  maskprep_k<<<B_ * 32, 256, 0, stream>>>(cell_masks, Mb, S);
  // big GEMMs (WMMA bf16)
  gemm_backbone_k<<<dim3(C_FEAT / BN, M_PATCH / BM), 256, 0, stream>>>(A_im2col, Wb_bf, b_backbone, fmT);
  gemm_roi_k<<<dim3(C_FEAT / 256, B_), 256, 0, stream>>>(Mb, fmT, S, cf32, cf16);
  gemm_mlp_k<<<dim3(H_DIM / 64, (B_ * K_CELLS) / 128), 256, 0, stream>>>(cf16, W1_bf, b1, H);
  // small heads + attention combine
  class_logits_k<<<B_ * K_CELLS, 32, 0, stream>>>(cf32, S, W_final, b_final, cl);
  weight_logits_k<<<B_ * K_CELLS, 32, 0, stream>>>(H, W2f, b2, wl);
  finalize_k<<<B_, 32, 0, stream>>>(wl, cl, cell_counts, W_final, b_final, fmT, out);
}
